// TransformerModule_46763603919125
// MI455X (gfx1250) — matrix operand order repaired and checked by reading, not compiled, not run on hardware
//
#include <hip/hip_runtime.h>

typedef __bf16 bf16_t;
typedef __attribute__((ext_vector_type(16))) __bf16 v16bf;
typedef __attribute__((ext_vector_type(8)))  __bf16 v8bf;
typedef __attribute__((ext_vector_type(8)))  float  v8f;

#define B_DIM 2048
#define S_LEN 100
#define F_IN  158
#define F_PAD 160
#define D_DIM 512
#define H_NUM 8
#define HD    64
#define MS    204800   // B*S

static __device__ inline v16bf cat8(v8bf lo, v8bf hi) {
  return __builtin_shufflevector(lo, hi, 0,1,2,3,4,5,6,7,8,9,10,11,12,13,14,15);
}

static __device__ inline v8f wmma_bf16(v16bf a, v16bf b, v8f c) {
  // (neg_a, A, neg_b, B, c_mod, C, reuse_a, reuse_b)
  return __builtin_amdgcn_wmma_f32_16x16x32_bf16(false, a, false, b, (short)0, c, false, false);
}

// A-fragment (16x32 bf16): lane L holds row m0+L%16; K = k0 + (L/16)*16 .. +15 contiguous.
static __device__ inline v16bf load_a_frag(const bf16_t* __restrict__ A, int lda,
                                           int m0, int k0, int lane) {
  int m   = m0 + (lane & 15);
  int klo = (lane >> 4) * 16;
  const bf16_t* p = A + (size_t)m * lda + k0 + klo;
  v8bf lo = *(const v8bf*)p;
  v8bf hi = *(const v8bf*)(p + 8);
  return cat8(lo, hi);
}

// B-fragment (32x16 bf16), B = W^T with W [N,K] row-major:
// lane L holds column n0+L%16 (= W row), K = k0 + (L/16)*16 .. +15 contiguous.
static __device__ inline v16bf load_b_frag(const bf16_t* __restrict__ W, int ldw,
                                           int n0, int k0, int lane) {
  int n  = n0 + (lane & 15);
  int kk = k0 + ((lane >> 4) << 4);
  return *(const v16bf*)(W + (size_t)n * ldw + kk);
}

// ---------------- prep kernels ----------------

__global__ void convert_pad_kernel(const float* __restrict__ in, bf16_t* __restrict__ out,
                                   int M, int K, int Mp, int Kp) {
  size_t i = (size_t)blockIdx.x * blockDim.x + threadIdx.x;
  size_t total = (size_t)Mp * Kp;
  if (i >= total) return;
  int r = (int)(i / Kp), c = (int)(i % Kp);
  float v = (r < M && c < K) ? in[(size_t)r * K + c] : 0.f;
  out[i] = (bf16_t)v;
}

__global__ void transpose_convert_kernel(const float* __restrict__ in, bf16_t* __restrict__ out, int N) {
  int i = blockIdx.x * blockDim.x + threadIdx.x;
  if (i >= N * N) return;
  int r = i / N, c = i % N;
  out[i] = (bf16_t)in[(size_t)c * N + r];   // out[r][c] = in[c][r]
}

__global__ void gather_last_kernel(const bf16_t* __restrict__ y2, bf16_t* __restrict__ y99) {
  int i = blockIdx.x * blockDim.x + threadIdx.x;
  if (i >= B_DIM * D_DIM) return;
  int b = i >> 9, e = i & 511;
  y99[i] = y2[((size_t)(b * S_LEN + S_LEN - 1)) * D_DIM + e];
}

// ---------------- generic WMMA GEMM: out = act(A @ W^T + bias [+ pe]) ----------------
// A [M,K] bf16, W [N,K] bf16 (both padded). Each wave computes NT consecutive
// 16x16 N-tiles for one M-tile, reusing the A fragment across them (register
// blocking over N -> 1 A-load feeds NT WMMAs).

template<int NT>
__global__ __launch_bounds__(256)
void gemm_bf16_kernel(const bf16_t* __restrict__ A, const bf16_t* __restrict__ W,
                      const float* __restrict__ bias, int nbias,
                      const float* __restrict__ pe, int peS, int relu,
                      bf16_t* __restrict__ outb, float* __restrict__ outf,
                      int M, int N, int K) {
  int lane = threadIdx.x & 31;
  int wave = threadIdx.x >> 5;
  long gtile = (long)blockIdx.x * 8 + wave;
  int tilesN  = N >> 4;
  int groupsN = tilesN / NT;
  long nGroups = (long)(M >> 4) * groupsN;
  if (gtile >= nGroups) return;
  int m0 = (int)(gtile / groupsN) << 4;
  int nb = (int)(gtile % groupsN) * (NT << 4);

  v8f acc[NT];
#pragma unroll
  for (int jn = 0; jn < NT; ++jn) acc[jn] = (v8f){0.f,0.f,0.f,0.f,0.f,0.f,0.f,0.f};

  for (int k0 = 0; k0 < K; k0 += 32) {
    v16bf a = load_a_frag(A, K, m0, k0, lane);
#pragma unroll
    for (int jn = 0; jn < NT; ++jn) {
      v16bf b = load_b_frag(W, K, nb + (jn << 4), k0, lane);
      acc[jn] = wmma_bf16(a, b, acc[jn]);
    }
  }
  int half = lane >> 4;
#pragma unroll
  for (int jn = 0; jn < NT; ++jn) {
    int n = nb + (jn << 4) + (lane & 15);
    float bv = (bias != nullptr && n < nbias) ? bias[n] : 0.f;
#pragma unroll
    for (int r = 0; r < 8; ++r) {
      int m = m0 + r + half * 8;
      float v = acc[jn][r] + bv;
      if (pe) v += pe[(size_t)(m % peS) * N + n];
      if (relu) v = v > 0.f ? v : 0.f;
      size_t idx = (size_t)m * N + n;
      if (outb) outb[idx] = (bf16_t)v;
      else      outf[idx] = v;
    }
  }
}

// ---------------- GEMM fused with out = g + LayerNorm(g), N fixed to 512 ----------------
// Block = 16 rows x full 512 cols; wave w owns N-tiles 4w..4w+3.

__global__ __launch_bounds__(256)
void gemm_ln_kernel(const bf16_t* __restrict__ A, const bf16_t* __restrict__ W,
                    const float* __restrict__ bias,
                    const float* __restrict__ ln_w, const float* __restrict__ ln_b,
                    bf16_t* __restrict__ out, int K) {
  __shared__ __align__(32) float g[16][D_DIM];   // 32 KB
  int lane = threadIdx.x & 31;
  int w    = threadIdx.x >> 5;
  int half = lane >> 4;
  int m0   = blockIdx.x << 4;

  v8f acc[4];
#pragma unroll
  for (int jn = 0; jn < 4; ++jn) acc[jn] = (v8f){0.f,0.f,0.f,0.f,0.f,0.f,0.f,0.f};

  for (int k0 = 0; k0 < K; k0 += 32) {
    v16bf a = load_a_frag(A, K, m0, k0, lane);
#pragma unroll
    for (int jn = 0; jn < 4; ++jn) {
      int n0 = ((w << 2) + jn) << 4;
      v16bf b = load_b_frag(W, K, n0, k0, lane);
      acc[jn] = wmma_bf16(a, b, acc[jn]);
    }
  }
#pragma unroll
  for (int jn = 0; jn < 4; ++jn) {
    int n = ((((w << 2) + jn) << 4)) + (lane & 15);
    float bv = bias ? bias[n] : 0.f;
#pragma unroll
    for (int r = 0; r < 8; ++r)
      g[r + half * 8][n] = acc[jn][r] + bv;
  }
  __syncthreads();

  int row  = threadIdx.x >> 4;   // 0..15
  int part = threadIdx.x & 15;   // 16 threads per row, 32 cols each
  float s1 = 0.f, s2 = 0.f;
  for (int c = 0; c < 32; ++c) {
    float x = g[row][part * 32 + c];
    s1 += x; s2 += x * x;
  }
  for (int off = 8; off >= 1; off >>= 1) {
    s1 += __shfl_xor(s1, off, 32);
    s2 += __shfl_xor(s2, off, 32);
  }
  float mean = s1 * (1.f / 512.f);
  float var  = s2 * (1.f / 512.f) - mean * mean;
  float rs   = rsqrtf(var + 1e-5f);
  size_t base = (size_t)(m0 + row) * D_DIM;
  for (int c = 0; c < 32; ++c) {
    int n = part * 32 + c;
    float x = g[row][n];
    float y = x + (x - mean) * rs * ln_w[n] + ln_b[n];
    out[base + n] = (bf16_t)y;
  }
}

// ---------------- fused attention: one block per (b,h) ----------------
// scores = q k^T / 8 (masked), softmax, attn = P V ; S=100 padded to 112/128.

__global__ __launch_bounds__(256)
void attention_kernel(const bf16_t* __restrict__ q, const bf16_t* __restrict__ kmat,
                      const bf16_t* __restrict__ v, bf16_t* __restrict__ attn) {
  int bh = blockIdx.x;
  int b  = bh >> 3;
  int h  = bh & 7;
  int tid = threadIdx.x;
  int lane = tid & 31, w = tid >> 5;

  __shared__ __align__(32) bf16_t VT[HD][128];      // V^T, 16 KB
  __shared__ __align__(32) bf16_t P[7][16][128];    // probs per wave, 28 KB

  size_t rowbase = (size_t)b * S_LEN;
  for (int i = tid; i < HD * 128; i += 256) {
    int d = i >> 7, kk = i & 127;
    float val = (kk < S_LEN) ? (float)v[(rowbase + kk) * D_DIM + h * HD + d] : 0.f;
    VT[d][kk] = (bf16_t)val;
  }
  for (int i = tid; i < 7 * 16 * 128; i += 256)
    ((bf16_t*)P)[i] = (bf16_t)0.f;
  __syncthreads();

  bool active = (w < 7);
  int half = lane >> 4;
  if (active) {
    int m0 = w * 16;
    v8f sc[7];
#pragma unroll
    for (int j = 0; j < 7; ++j) sc[j] = (v8f){0.f,0.f,0.f,0.f,0.f,0.f,0.f,0.f};

    int m = m0 + (lane & 15); if (m > S_LEN - 1) m = S_LEN - 1;   // clamp pad rows
    int klo = half * 16;
#pragma unroll
    for (int j = 0; j < 7; ++j) {
      int krow = j * 16 + (lane & 15); if (krow > S_LEN - 1) krow = S_LEN - 1;
      const bf16_t* brow = kmat + (rowbase + krow) * D_DIM + h * HD;
#pragma unroll
      for (int kk = 0; kk < HD; kk += 32) {
        const bf16_t* ap = q + (rowbase + m) * D_DIM + h * HD + kk + klo;
        v16bf a = cat8(*(const v8bf*)ap, *(const v8bf*)(ap + 8));
        v16bf bb = *(const v16bf*)(brow + kk + half * 16);
        sc[j] = wmma_bf16(a, bb, sc[j]);
      }
    }
    // masked softmax over key dim (cols spread across 7 tiles x 16-lane half)
#pragma unroll
    for (int r = 0; r < 8; ++r) {
      float vals[7];
      float mx = -3.0e38f;
#pragma unroll
      for (int j = 0; j < 7; ++j) {
        int n = j * 16 + (lane & 15);
        float s = sc[j][r] * 0.125f;            // 1/sqrt(64)
        if (n >= S_LEN) s = -3.0e38f;
        vals[j] = s;
        mx = fmaxf(mx, s);
      }
      for (int off = 8; off >= 1; off >>= 1) mx = fmaxf(mx, __shfl_xor(mx, off, 32));
      float sum = 0.f;
#pragma unroll
      for (int j = 0; j < 7; ++j) {
        float e = (vals[j] > -1e30f) ? __expf(vals[j] - mx) : 0.f;
        vals[j] = e; sum += e;
      }
      for (int off = 8; off >= 1; off >>= 1) sum += __shfl_xor(sum, off, 32);
      float inv = 1.f / sum;
      int mrow = r + half * 8;
#pragma unroll
      for (int j = 0; j < 7; ++j)
        P[w][mrow][j * 16 + (lane & 15)] = (bf16_t)(vals[j] * inv);
    }
  }
  __syncthreads();

  if (active) {
    int m0 = w * 16;
    v8f acc[4];
#pragma unroll
    for (int jn = 0; jn < 4; ++jn) acc[jn] = (v8f){0.f,0.f,0.f,0.f,0.f,0.f,0.f,0.f};
    int klo = half * 16;
#pragma unroll
    for (int kk = 0; kk < 128; kk += 32) {
      const bf16_t* ap = &P[w][lane & 15][kk + klo];
      v16bf a = cat8(*(const v8bf*)ap, *(const v8bf*)(ap + 8));
#pragma unroll
      for (int jn = 0; jn < 4; ++jn) {
        const bf16_t* bp = &VT[jn * 16 + (lane & 15)][kk + half * 16];
        v16bf bb = *(const v16bf*)bp;
        acc[jn] = wmma_bf16(a, bb, acc[jn]);
      }
    }
#pragma unroll
    for (int jn = 0; jn < 4; ++jn) {
      int n = jn * 16 + (lane & 15);
#pragma unroll
      for (int r = 0; r < 8; ++r) {
        int s = m0 + r + half * 8;
        if (s < S_LEN)
          attn[(rowbase + s) * D_DIM + h * HD + n] = (bf16_t)acc[jn][r];
      }
    }
  }
}

// ---------------- temporal pooling: lam = y2 . U, softmax, pooled . W_out ----------------

__global__ __launch_bounds__(128)
void pool_kernel(const bf16_t* __restrict__ y2, const float* __restrict__ U,
                 const float* __restrict__ W_out, const float* __restrict__ b_out,
                 float* __restrict__ out) {
  int b = blockIdx.x;
  int tid = threadIdx.x, lane = tid & 31, w = tid >> 5;
  __shared__ float lam[112];
  __shared__ float red[128];
  __shared__ float stats[1];
  const bf16_t* yb = y2 + (size_t)b * S_LEN * D_DIM;
  const float*  ub = U + (size_t)b * D_DIM;

  for (int s = w; s < S_LEN; s += 4) {
    float p = 0.f;
    for (int e = lane; e < D_DIM; e += 32) p += (float)yb[(size_t)s * D_DIM + e] * ub[e];
    for (int off = 16; off >= 1; off >>= 1) p += __shfl_xor(p, off, 32);
    if (lane == 0) lam[s] = p;
  }
  __syncthreads();
  if (tid == 0) {
    float mx = -3e38f;
    for (int s = 0; s < S_LEN; ++s) mx = fmaxf(mx, lam[s]);
    float sum = 0.f;
    for (int s = 0; s < S_LEN; ++s) { float e = __expf(lam[s] - mx); lam[s] = e; sum += e; }
    stats[0] = 1.f / sum;
  }
  __syncthreads();
  float inv = stats[0];
  float partial = 0.f;
  for (int e = tid; e < D_DIM; e += 128) {
    float acc = 0.f;
    for (int s = 0; s < S_LEN; ++s) acc += lam[s] * (float)yb[(size_t)s * D_DIM + e];
    partial += acc * inv * W_out[e];
  }
  red[tid] = partial;
  __syncthreads();
  for (int st = 64; st >= 1; st >>= 1) {
    if (tid < st) red[tid] += red[tid + st];
    __syncthreads();
  }
  if (tid == 0) out[b] = red[0] + b_out[0];
}

// ---------------- host-side orchestration ----------------

extern "C" void kernel_launch(void* const* d_in, const int* in_sizes, int n_in,
                              void* d_out, int out_size, void* d_ws, size_t ws_size,
                              hipStream_t stream) {
  (void)in_sizes; (void)n_in; (void)out_size; (void)ws_size;
  const float* src    = (const float*)d_in[0];
  const float* W_feat = (const float*)d_in[1];
  const float* b_feat = (const float*)d_in[2];
  const float* pe     = (const float*)d_in[3];
  const float* Wq = (const float*)d_in[4];  const float* bq = (const float*)d_in[5];
  const float* Wk = (const float*)d_in[6];  const float* bk = (const float*)d_in[7];
  const float* Wv = (const float*)d_in[8];  const float* bv = (const float*)d_in[9];
  const float* Wo = (const float*)d_in[10]; const float* bo = (const float*)d_in[11];
  const float* ln_w = (const float*)d_in[12];
  const float* ln_b = (const float*)d_in[13];
  const float* W1 = (const float*)d_in[14]; const float* b1 = (const float*)d_in[15];
  const float* W2 = (const float*)d_in[16]; const float* b2 = (const float*)d_in[17];
  const float* Wt = (const float*)d_in[18];
  const float* W_out = (const float*)d_in[19];
  const float* b_out = (const float*)d_in[20];
  float* out = (float*)d_out;

  char* ws = (char*)d_ws;
  size_t off = 0;
  auto take = [&](size_t bytes) -> char* {
    char* p = ws + off;
    off += (bytes + 255) & ~(size_t)255;
    return p;
  };
  bf16_t* srcbf  = (bf16_t*)take((size_t)MS * F_PAD * 2);
  bf16_t* xbuf   = (bf16_t*)take((size_t)MS * D_DIM * 2);
  bf16_t* qbuf   = (bf16_t*)take((size_t)MS * D_DIM * 2);
  bf16_t* kbuf   = (bf16_t*)take((size_t)MS * D_DIM * 2);
  bf16_t* vbuf   = (bf16_t*)take((size_t)MS * D_DIM * 2);
  bf16_t* wfeatbf = (bf16_t*)take((size_t)D_DIM * F_PAD * 2);
  bf16_t* wqbf   = (bf16_t*)take((size_t)D_DIM * D_DIM * 2);
  bf16_t* wkbf   = (bf16_t*)take((size_t)D_DIM * D_DIM * 2);
  bf16_t* wvbf   = (bf16_t*)take((size_t)D_DIM * D_DIM * 2);
  bf16_t* wobf   = (bf16_t*)take((size_t)D_DIM * D_DIM * 2);
  bf16_t* w1bf   = (bf16_t*)take((size_t)F_PAD * D_DIM * 2);
  bf16_t* w2bf   = (bf16_t*)take((size_t)D_DIM * F_PAD * 2);
  bf16_t* wtbf   = (bf16_t*)take((size_t)D_DIM * D_DIM * 2);
  bf16_t* wtTbf  = (bf16_t*)take((size_t)D_DIM * D_DIM * 2);
  bf16_t* y99    = (bf16_t*)take((size_t)B_DIM * D_DIM * 2);
  bf16_t* h99    = (bf16_t*)take((size_t)B_DIM * D_DIM * 2);
  float*  Ubuf   = (float*)take((size_t)B_DIM * D_DIM * 4);
  // buffer reuse (dead after their consumers):
  bf16_t* attn = xbuf;    // x dead after QKV
  bf16_t* y1   = qbuf;    // q dead after attention
  bf16_t* tbuf = srcbf;   // src_bf dead after feat projection
  bf16_t* y2   = kbuf;    // k dead after attention

  auto cvt = [&](const float* in, bf16_t* o, int M, int K, int Mp, int Kp) {
    size_t total = (size_t)Mp * Kp;
    int blocks = (int)((total + 255) / 256);
    convert_pad_kernel<<<dim3(blocks), dim3(256), 0, stream>>>(in, o, M, K, Mp, Kp);
  };
  // NT=4 register-blocked GEMM (N % 64 == 0)
  auto gemm4 = [&](const bf16_t* A, const bf16_t* W, const float* bias, int nbias,
                   const float* pe_p, int peS, int relu,
                   bf16_t* ob, float* of, int M, int N, int K) {
    long groups = (long)(M >> 4) * ((N >> 4) / 4);
    int blocks = (int)((groups + 7) / 8);
    gemm_bf16_kernel<4><<<dim3(blocks), dim3(256), 0, stream>>>(
        A, W, bias, nbias, pe_p, peS, relu, ob, of, M, N, K);
  };
  // NT=2 register-blocked GEMM (N % 32 == 0)
  auto gemm2 = [&](const bf16_t* A, const bf16_t* W, const float* bias, int nbias,
                   const float* pe_p, int peS, int relu,
                   bf16_t* ob, float* of, int M, int N, int K) {
    long groups = (long)(M >> 4) * ((N >> 4) / 2);
    int blocks = (int)((groups + 7) / 8);
    gemm_bf16_kernel<2><<<dim3(blocks), dim3(256), 0, stream>>>(
        A, W, bias, nbias, pe_p, peS, relu, ob, of, M, N, K);
  };

  // ---- prep: bf16 padded copies ----
  cvt(src,    srcbf,  MS, F_IN, MS, F_PAD);
  cvt(W_feat, wfeatbf, D_DIM, F_IN, D_DIM, F_PAD);
  cvt(Wq, wqbf, D_DIM, D_DIM, D_DIM, D_DIM);
  cvt(Wk, wkbf, D_DIM, D_DIM, D_DIM, D_DIM);
  cvt(Wv, wvbf, D_DIM, D_DIM, D_DIM, D_DIM);
  cvt(Wo, wobf, D_DIM, D_DIM, D_DIM, D_DIM);
  cvt(W1, w1bf, F_IN, D_DIM, F_PAD, D_DIM);
  cvt(W2, w2bf, D_DIM, F_IN, D_DIM, F_PAD);
  cvt(Wt, wtbf, D_DIM, D_DIM, D_DIM, D_DIM);
  transpose_convert_kernel<<<dim3((D_DIM * D_DIM + 255) / 256), dim3(256), 0, stream>>>(
      Wt, wtTbf, D_DIM);

  // ---- 1. feature projection + PE -> x ----
  gemm4(srcbf, wfeatbf, b_feat, D_DIM, pe, S_LEN, 0, xbuf, nullptr, MS, D_DIM, F_PAD);

  // ---- 2. QKV projections ----
  gemm4(xbuf, wqbf, bq, D_DIM, nullptr, 0, 0, qbuf, nullptr, MS, D_DIM, D_DIM);
  gemm4(xbuf, wkbf, bk, D_DIM, nullptr, 0, 0, kbuf, nullptr, MS, D_DIM, D_DIM);
  gemm4(xbuf, wvbf, bv, D_DIM, nullptr, 0, 0, vbuf, nullptr, MS, D_DIM, D_DIM);

  // ---- 3. fused attention (softmax in-registers, V^T + probs in LDS) ----
  attention_kernel<<<dim3(B_DIM * H_NUM), dim3(256), 0, stream>>>(qbuf, kbuf, vbuf, attn);

  // ---- 4. O-proj + addnorm#1 -> y1 ----
  gemm_ln_kernel<<<dim3(MS / 16), dim3(256), 0, stream>>>(attn, wobf, bo, ln_w, ln_b, y1, D_DIM);

  // ---- 5. FFN hidden (relu) -> t [MS,160] ----
  gemm2(y1, w1bf, b1, F_IN, nullptr, 0, 1, tbuf, nullptr, MS, F_PAD, D_DIM);

  // ---- 6. FFN out + addnorm#2 -> y2 ----
  gemm_ln_kernel<<<dim3(MS / 16), dim3(256), 0, stream>>>(tbuf, w2bf, b2, ln_w, ln_b, y2, F_PAD);

  // ---- 7. temporal pooling: lam[b,s] = y2[b,s] . (Wt^T Wt y2[b,99]) ----
  gather_last_kernel<<<dim3((B_DIM * D_DIM + 255) / 256), dim3(256), 0, stream>>>(y2, y99);
  gemm4(y99, wtbf,  nullptr, 0, nullptr, 0, 0, h99, nullptr, B_DIM, D_DIM, D_DIM); // H99 = Y99 Wt^T
  gemm4(h99, wtTbf, nullptr, 0, nullptr, 0, 0, nullptr, Ubuf, B_DIM, D_DIM, D_DIM); // U = H99 Wt
  pool_kernel<<<dim3(B_DIM), dim3(128), 0, stream>>>(y2, Ubuf, W_out, b_out, out);
}
